// LinearAttention_38087769981319
// MI455X (gfx1250) — compile-verified, matrix operand order checked by reading, not run on hardware
//
#include <hip/hip_runtime.h>

// ---------------- problem constants ----------------
#define B_SZ     16
#define C_DIM    512
#define T_LEN    4096
#define HEADS    8
#define DHEAD    64
#define HIDDEN   (HEADS * DHEAD)        // 512
#define QKV_ROWS (3 * HIDDEN)           // 1536
#define SCALE_Q  0.125f                 // 64^-0.5
#define GN_EPS   1e-5f

// ---------------- workspace layout (bytes) ----------------
// total required ~341 MB
#define OFF_WQKV  ((size_t)0)                    // 1536*512*2        = 1,572,864
#define OFF_WOUT  ((size_t)0x180000)             // 512*512*2         =   524,288
#define OFF_STATS ((size_t)0x200000)             // 16*2*4            =       128
#define OFF_CTX   ((size_t)0x210000)             // 16*8*64*64*4      = 2,097,152
#define OFF_XBF   ((size_t)0x500000)             // 16*512*4096*2     = 67,108,864
#define OFF_QKV   ((size_t)0x4500000)            // 16*1536*4096*2    = 201,326,592
#define OFF_QT    ((size_t)0x10500000)           // 16*8*4096*64*2    = 67,108,864
#define OFF_ATTN  OFF_QKV                        // alias: qkv dead after attn is produced

// ---------------- WMMA / TDM types ----------------
typedef __attribute__((ext_vector_type(16))) __bf16 v16bf;
typedef __attribute__((ext_vector_type(8)))  float  v8f;
typedef __attribute__((ext_vector_type(4)))  unsigned int v4u;
typedef __attribute__((ext_vector_type(8)))  int    v8i;
typedef __attribute__((ext_vector_type(4)))  int    v4i;

union Frag {
  v16bf bf;
  unsigned int u[8];
};

static __device__ __forceinline__ v8f zero8() {
  v8f z = {0.f, 0.f, 0.f, 0.f, 0.f, 0.f, 0.f, 0.f};
  return z;
}

static __device__ __forceinline__ unsigned short f2bf(float f) {
  unsigned int u = __float_as_uint(f);
  unsigned int r = u + 0x7FFFu + ((u >> 16) & 1u);   // round-to-nearest-even
  return (unsigned short)(r >> 16);
}
static __device__ __forceinline__ float bf2f(unsigned short s) {
  return __uint_as_float(((unsigned int)s) << 16);
}
static __device__ __forceinline__ unsigned int pack2(float a, float b) {
  return (unsigned int)f2bf(a) | ((unsigned int)f2bf(b) << 16);
}

static __device__ __forceinline__ v8f wmma_bf16(const Frag& a, const Frag& b, v8f c) {
  // D = A(16x32 bf16) x B(32x16 bf16) + C(16x16 f32)
  return __builtin_amdgcn_wmma_f32_16x16x32_bf16(
      /*neg_a=*/false, a.bf, /*neg_b=*/false, b.bf,
      /*c_mod=*/(short)0, c, /*reuse_a=*/false, /*reuse_b=*/false);
}

// ---- Tensor Data Mover: DMA a 128x32 bf16 tile (rows of a [Mdim,Kdim]
// row-major matrix) into LDS, inserting 1 dword of padding after every 16
// dwords so the LDS image has pitch 34 halfwords (matches lA[][KT+2]).
// D# fields per CDNA5 ISA 8.3/8.4.  Issued by one wave; TENSORcnt-tracked.
// This toolchain exposes the 6-arg builtin:
//   (uint32x4 g0, int32x8 g1, int32x4 g2, int32x4 g3, int32x8, i32 cpol)
static __device__ __forceinline__ void tdm_load_tile_A(
    const unsigned short* gsrc, unsigned int lds_byte_off, int Mdim, int Kdim) {
  unsigned long long ga = (unsigned long long)(const void*)gsrc;
  v4u g0;
  g0[0] = 1u;                                                       // count=1 (valid user D#)
  g0[1] = lds_byte_off;                                             // lds_addr
  g0[2] = (unsigned int)(ga & 0xFFFFFFFFull);                       // global_addr[31:0]
  g0[3] = (unsigned int)((ga >> 32) & 0x01FFFFFFull) | (2u << 30);  // addr[56:32] | type=2
  v8i g1;
  g1[0] = (int)((1u << 16)            // data_size = 1 -> 2 bytes
              | (1u << 20)            // pad_enable
              | (3u << 22)            // pad_interval = 16 dwords (one 32-elem row)
              | (0u << 25));          // pad_amount = 1 dword
  g1[1] = (int)(((unsigned)Kdim & 0xFFFFu) << 16);                        // tensor_dim0[15:0]
  g1[2] = (int)((((unsigned)Kdim >> 16) & 0xFFFFu)
              | (((unsigned)Mdim & 0xFFFFu) << 16));                      // dim0 hi | dim1 lo
  g1[3] = (int)((((unsigned)Mdim >> 16) & 0xFFFFu) | (32u << 16));        // dim1 hi | tile_dim0=32
  g1[4] = (int)128u;                                                      // tile_dim1=128, tile_dim2=0
  g1[5] = (int)Kdim;                                                      // tensor_dim0_stride lo32
  g1[6] = 0;                                                              // stride hi | dim1_stride lo
  g1[7] = 0;
  v4i gz4 = {0, 0, 0, 0};
  v8i gz8 = {0, 0, 0, 0, 0, 0, 0, 0};
  __builtin_amdgcn_tensor_load_to_lds(g0, g1, gz4, gz4, gz8, 0);
}

// Load one 16x32 bf16 fragment directly from a row-major global matrix
// (row stride 4096 elements). p must point at element (m, kbase) and be
// 16-byte aligned; elements kbase..kbase+7 and kbase+16..kbase+23 are taken.
static __device__ __forceinline__ Frag load_frag_g4096(const unsigned short* p) {
  Frag f;
  const uint4* q = (const uint4*)p;
  uint4 lo = q[0];
  uint4 hi = q[2];   // +16 bf16 elements
  f.u[0] = lo.x; f.u[1] = lo.y; f.u[2] = lo.z; f.u[3] = lo.w;
  f.u[4] = hi.x; f.u[5] = hi.y; f.u[6] = hi.z; f.u[7] = hi.w;
  return f;
}

// ---------------- elementwise fp32 -> bf16 ----------------
__global__ void f32_to_bf16_kernel(const float* __restrict__ in,
                                   unsigned short* __restrict__ out,
                                   long long n) {
  long long i = (long long)blockIdx.x * blockDim.x + threadIdx.x;
  long long stride = (long long)gridDim.x * blockDim.x;
  for (; i < n; i += stride) out[i] = f2bf(in[i]);
}

__global__ void zero_stats_kernel(float* __restrict__ stats) {
  if (threadIdx.x < 32) stats[threadIdx.x] = 0.f;
}

// ---------------- tiled WMMA GEMM: C[M,N] = A[M,K] * B[K,N] ----------------
// A, B bf16; C either bf16 (Cb) or f32 (Cf, with bias + GroupNorm stats).
// Block tile 128x64, KT=32, 256 threads (8 waves, each wave owns 32x32).
// A tiles DMA'd by the Tensor Data Mover (wave 0), B tiles software-pipelined
// through registers with an LDS transpose; double-buffered, 1 barrier/K-step.
#define MT 128
#define NT 64
#define KT 32

__global__ __launch_bounds__(256) void gemm_bf16_kernel(
    const unsigned short* __restrict__ A,
    const unsigned short* __restrict__ Bg,
    unsigned short* __restrict__ Cb,
    float* __restrict__ Cf,
    const float* __restrict__ bias,
    float* __restrict__ statsBase,
    int M, int K, int N) {
  __shared__ unsigned short lA[2][MT][KT + 2];   // pitch 34 = 17 dwords (odd -> bank friendly)
  __shared__ unsigned short lB[2][NT][KT + 2];   // stored [N][K] (B transposed)
  __shared__ float red[256];

  const int z = blockIdx.z;
  Bg += (size_t)z * K * N;
  if (Cb) Cb += (size_t)z * M * N;
  if (Cf) Cf += (size_t)z * M * N;

  const int tid  = threadIdx.x;
  const int lane = tid & 31;
  const int w    = tid >> 5;
  const int wm   = w & 3;          // 4 wave-rows  * 32 = 128
  const int wn   = w >> 2;         // 2 wave-cols  * 32 = 64
  const int m0   = blockIdx.y * MT;
  const int n0   = blockIdx.x * NT;
  const int fr   = lane & 15;
  const int kb8  = (lane >> 4) * 8;

  // B-tile pipeline helpers: each thread owns 4 dwords (8 bf16) of the
  // 32x64 tile; global reads coalesced along n, LDS stores transposed.
  const int bn = (tid & 31) * 2;       // n within tile (0..62, even)
  const int bk0 = tid >> 5;            // k for chunk 0; chunks step k by 8
  unsigned int bv[4];

  v8f acc[2][2];
  acc[0][0] = zero8(); acc[0][1] = zero8();
  acc[1][0] = zero8(); acc[1][1] = zero8();

  // ---- prologue: stage tile kt=0 into buffer 0 ----
  if (tid < 32)
    tdm_load_tile_A(A + (size_t)m0 * K, (unsigned int)(size_t)&lA[0][0][0], M, K);
  #pragma unroll
  for (int j = 0; j < 4; j++)
    bv[j] = *(const unsigned int*)(Bg + (size_t)(bk0 + j * 8) * N + n0 + bn);
  #pragma unroll
  for (int j = 0; j < 4; j++) {
    lB[0][bn][bk0 + j * 8]     = (unsigned short)(bv[j] & 0xFFFFu);
    lB[0][bn + 1][bk0 + j * 8] = (unsigned short)(bv[j] >> 16);
  }
  if (tid < 32) __builtin_amdgcn_s_wait_tensorcnt(0);
  __syncthreads();

  int buf = 0;
  for (int kt = 0; kt < K; kt += KT) {
    const int nxt = buf ^ 1;
    const bool more = (kt + KT) < K;

    if (more) {   // kick off next tile while we compute this one
      if (tid < 32)
        tdm_load_tile_A(A + (size_t)m0 * K + (kt + KT),
                        (unsigned int)(size_t)&lA[nxt][0][0], M, K);
      #pragma unroll
      for (int j = 0; j < 4; j++)
        bv[j] = *(const unsigned int*)(Bg + (size_t)(kt + KT + bk0 + j * 8) * N + n0 + bn);
    }

    Frag a[2], b[2];
    #pragma unroll
    for (int i = 0; i < 2; i++) {
      const unsigned int* p = (const unsigned int*)&lA[buf][wm * 32 + i * 16 + fr][kb8];
      #pragma unroll
      for (int j = 0; j < 4; j++) { a[i].u[j] = p[j]; a[i].u[4 + j] = p[8 + j]; }
    }
    #pragma unroll
    for (int i = 0; i < 2; i++) {
      const unsigned int* p = (const unsigned int*)&lB[buf][wn * 32 + i * 16 + fr][kb8];
      #pragma unroll
      for (int j = 0; j < 4; j++) { b[i].u[j] = p[j]; b[i].u[4 + j] = p[8 + j]; }
    }
    acc[0][0] = wmma_bf16(a[0], b[0], acc[0][0]);
    acc[0][1] = wmma_bf16(a[0], b[1], acc[0][1]);
    acc[1][0] = wmma_bf16(a[1], b[0], acc[1][0]);
    acc[1][1] = wmma_bf16(a[1], b[1], acc[1][1]);

    if (more) {   // drain pipelined B into the next buffer
      #pragma unroll
      for (int j = 0; j < 4; j++) {
        lB[nxt][bn][bk0 + j * 8]     = (unsigned short)(bv[j] & 0xFFFFu);
        lB[nxt][bn + 1][bk0 + j * 8] = (unsigned short)(bv[j] >> 16);
      }
      if (tid < 32) __builtin_amdgcn_s_wait_tensorcnt(0);
    }
    __syncthreads();
    buf = nxt;
  }

  // C layout: VGPR r, lanes 0-15 -> M=r, N=lane; lanes 16-31 -> M=r+8, N=lane-16
  float s = 0.f, ss = 0.f;
  #pragma unroll
  for (int i = 0; i < 2; i++)
    #pragma unroll
    for (int j = 0; j < 2; j++)
      #pragma unroll
      for (int r = 0; r < 8; r++) {
        int row = m0 + wm * 32 + i * 16 + r + ((lane >> 4) << 3);
        int col = n0 + wn * 32 + j * 16 + (lane & 15);
        float v = acc[i][j][r];
        if (bias) v += bias[row];
        if (Cf) Cf[(size_t)row * N + col] = v;
        else    Cb[(size_t)row * N + col] = f2bf(v);
        s += v; ss += v * v;
      }

  if (statsBase) {   // per-batch sum / sumsq for GroupNorm
    red[tid] = s;  __syncthreads();
    for (int st = 128; st > 0; st >>= 1) { if (tid < st) red[tid] += red[tid + st]; __syncthreads(); }
    if (tid == 0) atomicAdd(statsBase + 2 * z + 0, red[0]);
    __syncthreads();
    red[tid] = ss; __syncthreads();
    for (int st = 128; st > 0; st >>= 1) { if (tid < st) red[tid] += red[tid + st]; __syncthreads(); }
    if (tid == 0) atomicAdd(statsBase + 2 * z + 1, red[0]);
  }
}

// ---------------- k softmax over t (in-place, bf16 rows of 4096) ----------------
__global__ __launch_bounds__(256) void ksoftmax_kernel(unsigned short* __restrict__ qkv) {
  __shared__ float red[256];
  const int row = blockIdx.x;          // 0 .. 16*512-1
  const int b = row >> 9, r = row & 511;
  unsigned short* p = qkv + ((size_t)b * QKV_ROWS + HIDDEN + r) * T_LEN;
  const int tid = threadIdx.x;

  float v[16];
  float m = -3.4e38f;
  #pragma unroll
  for (int j = 0; j < 16; j++) { v[j] = bf2f(p[tid + j * 256]); m = fmaxf(m, v[j]); }
  red[tid] = m; __syncthreads();
  for (int st = 128; st > 0; st >>= 1) { if (tid < st) red[tid] = fmaxf(red[tid], red[tid + st]); __syncthreads(); }
  const float rmax = red[0]; __syncthreads();

  float s = 0.f;
  #pragma unroll
  for (int j = 0; j < 16; j++) { v[j] = __expf(v[j] - rmax); s += v[j]; }
  red[tid] = s; __syncthreads();
  for (int st = 128; st > 0; st >>= 1) { if (tid < st) red[tid] += red[tid + st]; __syncthreads(); }
  const float inv = 1.0f / red[0];
  #pragma unroll
  for (int j = 0; j < 16; j++) p[tid + j * 256] = f2bf(v[j] * inv);
}

// ---- q softmax over d (64, strided) -> qT[b,h,t,64] bf16, scaled by 1/sqrt(d) ----
__global__ __launch_bounds__(256) void qsoftmax_kernel(const unsigned short* __restrict__ qkv,
                                                       unsigned short* __restrict__ qT) {
  const int g = blockIdx.x * 256 + threadIdx.x;        // 0 .. 16*8*4096-1
  const int b = g >> 15, h = (g >> 12) & 7, t = g & 4095;
  const unsigned short* p = qkv + ((size_t)b * QKV_ROWS + (size_t)h * DHEAD) * T_LEN + t;
  float m = -3.4e38f;
  for (int d = 0; d < DHEAD; d++) m = fmaxf(m, bf2f(p[(size_t)d * T_LEN]));
  float s = 0.f;
  for (int d = 0; d < DHEAD; d++) s += __expf(bf2f(p[(size_t)d * T_LEN]) - m);
  const float inv = SCALE_Q / s;
  unsigned short* o = qT + (size_t)g * DHEAD;
  for (int d = 0; d < DHEAD; d++)
    o[d] = f2bf(__expf(bf2f(p[(size_t)d * T_LEN]) - m) * inv);
}

// ---------------- context[b,h,64,64] = k_sm[64,4096] * v[64,4096]^T ----------------
__global__ __launch_bounds__(256) void ctx_kernel(const unsigned short* __restrict__ qkv,
                                                  float* __restrict__ ctxb) {
  const int bh = blockIdx.x;
  const int b = bh >> 3, h = bh & 7;
  const unsigned short* Kp = qkv + ((size_t)b * QKV_ROWS +     HIDDEN + (size_t)h * DHEAD) * T_LEN;
  const unsigned short* Vp = qkv + ((size_t)b * QKV_ROWS + 2 * HIDDEN + (size_t)h * DHEAD) * T_LEN;
  const int tid = threadIdx.x, lane = tid & 31, w = tid >> 5;
  const int mi = w >> 1;               // 4 m-tiles of 16
  const int nb = (w & 1) * 32;         // each wave: 2 n-tiles of 16
  const int fr = lane & 15, kb8 = (lane >> 4) * 8;

  v8f acc0 = zero8(), acc1 = zero8();
  for (int kt = 0; kt < T_LEN; kt += 32) {
    Frag a  = load_frag_g4096(Kp + (size_t)(mi * 16 + fr) * T_LEN + kt + kb8);
    Frag b0 = load_frag_g4096(Vp + (size_t)(nb + fr)      * T_LEN + kt + kb8);
    Frag b1 = load_frag_g4096(Vp + (size_t)(nb + 16 + fr) * T_LEN + kt + kb8);
    acc0 = wmma_bf16(a, b0, acc0);
    acc1 = wmma_bf16(a, b1, acc1);
  }
  #pragma unroll
  for (int r = 0; r < 8; r++) {
    int d = mi * 16 + r + ((lane >> 4) << 3);
    int e0 = nb + (lane & 15);
    ctxb[((size_t)bh * 64 + d) * 64 + e0]      = acc0[r];
    ctxb[((size_t)bh * 64 + d) * 64 + e0 + 16] = acc1[r];
  }
}

// ---- attn[b,c=h*64+e,t] = sum_d ctx[d,e] * q[d,t]  (A = ctx^T, B = qT) ----
__global__ __launch_bounds__(256) void attn_kernel(const float* __restrict__ ctxb,
                                                   const unsigned short* __restrict__ qT,
                                                   unsigned short* __restrict__ attn) {
  const int bh = blockIdx.y;
  const int b = bh >> 3, h = bh & 7;
  const int t0 = blockIdx.x * 128;
  const float* Ctx = ctxb + (size_t)bh * 64 * 64;
  const unsigned short* Qt = qT + (size_t)bh * T_LEN * DHEAD;
  unsigned short* Out = attn + ((size_t)b * C_DIM + (size_t)h * DHEAD) * T_LEN;

  const int tid = threadIdx.x, lane = tid & 31, w = tid >> 5;
  const int wmi = w & 1;               // 2 * 32 = 64 rows (e)
  const int wni = w >> 1;              // 4 * 32 = 128 cols (t)
  const int fr = lane & 15, kb8 = (lane >> 4) * 8;

  v8f acc[2][2];
  acc[0][0] = zero8(); acc[0][1] = zero8();
  acc[1][0] = zero8(); acc[1][1] = zero8();

  for (int ks = 0; ks < 2; ks++) {     // K = 64 = 2 * 32
    const int d0 = ks * 32 + kb8;
    Frag a[2];
    #pragma unroll
    for (int i = 0; i < 2; i++) {
      int m = wmi * 32 + i * 16 + fr;  // e index (A = ctx^T)
      #pragma unroll
      for (int j = 0; j < 4; j++) {
        a[i].u[j]     = pack2(Ctx[(d0 + 2 * j) * 64 + m],      Ctx[(d0 + 2 * j + 1) * 64 + m]);
        a[i].u[4 + j] = pack2(Ctx[(d0 + 16 + 2 * j) * 64 + m], Ctx[(d0 + 17 + 2 * j) * 64 + m]);
      }
    }
    Frag bfr[2];
    #pragma unroll
    for (int j2 = 0; j2 < 2; j2++) {
      int t = t0 + wni * 32 + j2 * 16 + fr;
      const uint4* p = (const uint4*)(Qt + (size_t)t * DHEAD + ks * 32 + kb8);
      uint4 lo = p[0], hi = p[2];
      bfr[j2].u[0] = lo.x; bfr[j2].u[1] = lo.y; bfr[j2].u[2] = lo.z; bfr[j2].u[3] = lo.w;
      bfr[j2].u[4] = hi.x; bfr[j2].u[5] = hi.y; bfr[j2].u[6] = hi.z; bfr[j2].u[7] = hi.w;
    }
    acc[0][0] = wmma_bf16(a[0], bfr[0], acc[0][0]);
    acc[0][1] = wmma_bf16(a[0], bfr[1], acc[0][1]);
    acc[1][0] = wmma_bf16(a[1], bfr[0], acc[1][0]);
    acc[1][1] = wmma_bf16(a[1], bfr[1], acc[1][1]);
  }
  #pragma unroll
  for (int i = 0; i < 2; i++)
    #pragma unroll
    for (int j = 0; j < 2; j++)
      #pragma unroll
      for (int r = 0; r < 8; r++) {
        int e = wmi * 32 + i * 16 + r + ((lane >> 4) << 3);
        int t = t0 + wni * 32 + j * 16 + (lane & 15);
        Out[(size_t)e * T_LEN + t] = f2bf(acc[i][j][r]);
      }
}

// ---------------- GroupNorm apply ----------------
__global__ void gn_kernel(float* __restrict__ out, const float* __restrict__ stats,
                          const float* __restrict__ gw, const float* __restrict__ gb) {
  const float invN = 1.0f / (float)((size_t)C_DIM * T_LEN);
  const size_t total = (size_t)B_SZ * C_DIM * T_LEN;
  size_t i = (size_t)blockIdx.x * blockDim.x + threadIdx.x;
  const size_t stride = (size_t)gridDim.x * blockDim.x;
  for (; i < total; i += stride) {
    int b = (int)(i >> 21);
    int c = (int)((i >> 12) & 511);
    float mean = stats[2 * b] * invN;
    float var  = stats[2 * b + 1] * invN - mean * mean;
    float inv  = rsqrtf(var + GN_EPS);
    out[i] = (out[i] - mean) * inv * gw[c] + gb[c];
  }
}

// ---------------- host launch ----------------
extern "C" void kernel_launch(void* const* d_in, const int* in_sizes, int n_in,
                              void* d_out, int out_size, void* d_ws, size_t ws_size,
                              hipStream_t stream) {
  const float* x     = (const float*)d_in[0];
  const float* Wqkv  = (const float*)d_in[1];
  const float* Wout  = (const float*)d_in[2];
  const float* b_out = (const float*)d_in[3];
  const float* gn_w  = (const float*)d_in[4];
  const float* gn_b  = (const float*)d_in[5];
  float* out = (float*)d_out;

  char* ws = (char*)d_ws;
  unsigned short* wqkv_bf = (unsigned short*)(ws + OFF_WQKV);
  unsigned short* wout_bf = (unsigned short*)(ws + OFF_WOUT);
  float*          stats   = (float*)(ws + OFF_STATS);
  float*          ctxb    = (float*)(ws + OFF_CTX);
  unsigned short* x_bf    = (unsigned short*)(ws + OFF_XBF);
  unsigned short* qkv_bf  = (unsigned short*)(ws + OFF_QKV);
  unsigned short* qT      = (unsigned short*)(ws + OFF_QT);
  unsigned short* attn    = (unsigned short*)(ws + OFF_ATTN);

  // 0) precision staging
  f32_to_bf16_kernel<<<3072, 256, 0, stream>>>(Wqkv, wqkv_bf, (long long)QKV_ROWS * C_DIM);
  f32_to_bf16_kernel<<<1024, 256, 0, stream>>>(Wout, wout_bf, (long long)C_DIM * HIDDEN);
  f32_to_bf16_kernel<<<8192, 256, 0, stream>>>(x, x_bf, (long long)B_SZ * C_DIM * T_LEN);
  zero_stats_kernel<<<1, 32, 0, stream>>>(stats);

  // 1) qkv = W_qkv @ x   (per batch: [1536,512] x [512,4096])
  gemm_bf16_kernel<<<dim3(T_LEN / NT, QKV_ROWS / MT, B_SZ), 256, 0, stream>>>(
      wqkv_bf, x_bf, qkv_bf, nullptr, nullptr, nullptr, QKV_ROWS, C_DIM, T_LEN);

  // 2) softmaxes
  ksoftmax_kernel<<<B_SZ * HIDDEN, 256, 0, stream>>>(qkv_bf);
  qsoftmax_kernel<<<(B_SZ * HEADS * T_LEN) / 256, 256, 0, stream>>>(qkv_bf, qT);

  // 3) context = k_sm @ v^T  ([64,4096] x [4096,64] per head)
  ctx_kernel<<<B_SZ * HEADS, 256, 0, stream>>>(qkv_bf, ctxb);

  // 4) attn = ctx^T @ q_sm  ([64,64] x [64,4096] per head) -> [b,512,4096] bf16
  attn_kernel<<<dim3(T_LEN / 128, B_SZ * HEADS), 256, 0, stream>>>(ctxb, qT, attn);

  // 5) out = W_out @ attn + b_out, with GroupNorm stats
  gemm_bf16_kernel<<<dim3(T_LEN / NT, C_DIM / MT, B_SZ), 256, 0, stream>>>(
      wout_bf, attn, nullptr, out, b_out, stats, C_DIM, C_DIM, T_LEN);

  // 6) GroupNorm apply
  gn_kernel<<<16384, 256, 0, stream>>>(out, stats, gn_w, gn_b);
}
